// GATtransformer_79920751444362
// MI455X (gfx1250) — matrix operand order repaired and checked by reading, not compiled, not run on hardware
//
#include <hip/hip_runtime.h>

// ---------------- problem constants (from reference) ----------------
#define N_NODES 4096   // G*S
#define DM      256    // d_model
#define HG      4      // GAT heads
#define HDM     1024   // HG * DM
#define EE      32768  // edges
#define EDIM    6
#define GN      32     // graphs
#define SN      128    // nodes per graph
#define NLAY    2

typedef __attribute__((ext_vector_type(16))) __bf16 v16bf;
typedef __attribute__((ext_vector_type(8)))  __bf16 v8bf;
typedef __attribute__((ext_vector_type(8)))  float  v8f;
typedef int v4i __attribute__((vector_size(16)));

#if __has_builtin(__builtin_amdgcn_global_load_async_to_lds_b128)
#define ASYNC_LDS 1
#endif

__device__ __forceinline__ void wait_async_lds() {
#ifdef ASYNC_LDS
#if __has_builtin(__builtin_amdgcn_s_wait_asynccnt)
  __builtin_amdgcn_s_wait_asynccnt(0);
#else
  asm volatile("s_wait_asynccnt 0x0" ::: "memory");
#endif
#endif
}

__device__ __forceinline__ v16bf pack16(v8bf lo, v8bf hi) {
  v16bf r;
#pragma unroll
  for (int i = 0; i < 8; ++i) { r[i] = lo[i]; r[8 + i] = hi[i]; }
  return r;
}

__device__ __forceinline__ v8f wmma_bf(v16bf a, v16bf b, v8f c) {
  return __builtin_amdgcn_wmma_f32_16x16x32_bf16(false, a, false, b, (short)0, c,
                                                 false, false);
}

__device__ __forceinline__ float gelu_f(float x) {
  return 0.5f * x * (1.0f + erff(x * 0.7071067811865476f));
}

// monotonic float<->uint encoding for atomic max on floats
__device__ __forceinline__ unsigned fenc(float f) {
  unsigned u = __float_as_uint(f);
  return (u & 0x80000000u) ? ~u : (u | 0x80000000u);
}
__device__ __forceinline__ float fdec(unsigned u) {
  return (u & 0x80000000u) ? __uint_as_float(u & 0x7fffffffu) : __uint_as_float(~u);
}
#define FENC_NEG_INF 0x007FFFFFu  // fenc(-inf)

// ---------------- fp32 -> bf16 convert ----------------
__global__ void k_cvt(const float* __restrict__ in, __bf16* __restrict__ out, int n) {
  for (int i = blockIdx.x * blockDim.x + threadIdx.x; i < n;
       i += gridDim.x * blockDim.x)
    out[i] = (__bf16)in[i];
}

// ---------------- embedding: h = LN(x@W+b)*g+bb + pos[s] ----------------
__global__ __launch_bounds__(256) void k_embed(
    const float* __restrict__ x, const float* __restrict__ W,
    const float* __restrict__ b, const float* __restrict__ g,
    const float* __restrict__ bb, const float* __restrict__ pos,
    float* __restrict__ h, __bf16* __restrict__ hbf) {
  int n = blockIdx.x, t = threadIdx.x;
  const float* xp = x + n * 16;
  float acc = b[t];
#pragma unroll
  for (int j = 0; j < 16; ++j) acc += xp[j] * W[j * DM + t];
  __shared__ float red[DM];
  red[t] = acc; __syncthreads();
  for (int s = 128; s > 0; s >>= 1) { if (t < s) red[t] += red[t + s]; __syncthreads(); }
  float mean = red[0] * (1.0f / DM); __syncthreads();
  float d = acc - mean;
  red[t] = d * d; __syncthreads();
  for (int s = 128; s > 0; s >>= 1) { if (t < s) red[t] += red[t + s]; __syncthreads(); }
  float var = red[0] * (1.0f / DM);
  float v = d * rsqrtf(var + 1e-5f) * g[t] + bb[t] + pos[(n % SN) * DM + t];
  h[n * DM + t] = v;
  hbf[n * DM + t] = (__bf16)v;
}

// ---------------- generic LayerNorm over DM ----------------
__global__ __launch_bounds__(256) void k_ln(
    const float* __restrict__ in, const float* __restrict__ g,
    const float* __restrict__ b, float* __restrict__ out) {
  int n = blockIdx.x, t = threadIdx.x;
  float v = in[n * DM + t];
  __shared__ float red[DM];
  red[t] = v; __syncthreads();
  for (int s = 128; s > 0; s >>= 1) { if (t < s) red[t] += red[t + s]; __syncthreads(); }
  float mean = red[0] * (1.0f / DM); __syncthreads();
  float d = v - mean;
  red[t] = d * d; __syncthreads();
  for (int s = 128; s > 0; s >>= 1) { if (t < s) red[t] += red[t + s]; __syncthreads(); }
  float var = red[0] * (1.0f / DM);
  out[n * DM + t] = d * rsqrtf(var + 1e-5f) * g[t] + b[t];
}

// ---------------- WMMA bf16 GEMM: C[M,Nn] = A[M,K]@B[K,Nn] + bias ----------------
// 256 threads (8 waves), block tile 128x128, wave tile 32x64 (2x4 wmma accum)
// Epilogue specialized at compile time: ACT (0/1=relu), RESID, WC (f32 out), WBF (bf16 out)
template <int ACT, bool RESID, bool WC, bool WBF>
__global__ __launch_bounds__(256) void k_gemm(
    const __bf16* __restrict__ A, const __bf16* __restrict__ B,
    const float* __restrict__ bias, const float* __restrict__ resid,
    float* __restrict__ C, __bf16* __restrict__ Cbf, int M, int Nn, int K) {
  __shared__ __bf16 lA[128 * 40];
  __shared__ __bf16 lB[128 * 40];  // B stored transposed: [n][k]
  const int bm = blockIdx.y * 128, bn = blockIdx.x * 128;
  const int t = threadIdx.x;
  const int wave = t >> 5, lane = t & 31;
  const int hl = lane >> 4, mn = lane & 15;
  const int wr = (wave >> 1) * 32;  // wave row offset: 0/32/64/96
  const int wc = (wave & 1) * 64;   // wave col offset: 0/64
  v8f acc[2][4];
#pragma unroll
  for (int a0 = 0; a0 < 2; ++a0)
#pragma unroll
    for (int b0 = 0; b0 < 4; ++b0)
#pragma unroll
      for (int i = 0; i < 8; ++i) acc[a0][b0][i] = 0.0f;

  const int arow = t >> 1, aseg = (t & 1) * 16;   // A: 128 rows x 32 k
  const int bkrow = t >> 3, bseg = (t & 7) * 16;  // B: 32 k-rows x 128 n

  for (int kk = 0; kk < K; kk += 32) {
    const __bf16* ag = A + (size_t)(bm + arow) * K + kk + aseg;
    const __bf16* bg = B + (size_t)(kk + bkrow) * Nn + bn + bseg;
#ifndef ASYNC_LDS
    v8bf a0v = *(const v8bf*)ag;
    v8bf a1v = *(const v8bf*)(ag + 8);
#endif
    v8bf b0v = *(const v8bf*)bg;
    v8bf b1v = *(const v8bf*)(bg + 8);
    if (kk + 32 < K) {
      __builtin_prefetch(ag + 32, 0, 1);
      __builtin_prefetch(bg + (size_t)32 * Nn, 0, 1);
    }
    __syncthreads();
#ifdef ASYNC_LDS
    // CDNA5 async DMA: global -> LDS, tracked by ASYNCcnt
    __builtin_amdgcn_global_load_async_to_lds_b128(
        (v4i*)ag,
        (__attribute__((address_space(3))) v4i*)&lA[arow * 40 + aseg], 0, 0);
    __builtin_amdgcn_global_load_async_to_lds_b128(
        (v4i*)(ag + 8),
        (__attribute__((address_space(3))) v4i*)&lA[arow * 40 + aseg + 8], 0, 0);
#else
    *(v8bf*)(&lA[arow * 40 + aseg]) = a0v;
    *(v8bf*)(&lA[arow * 40 + aseg + 8]) = a1v;
#endif
#pragma unroll
    for (int i = 0; i < 8; ++i) {
      lB[(bseg + i) * 40 + bkrow] = b0v[i];
      lB[(bseg + 8 + i) * 40 + bkrow] = b1v[i];
    }
    wait_async_lds();
    __syncthreads();
    // A fragments: lane row = mn, elems K = {0..7,16..23}+8*hl per ISA layout
    v16bf af[2];
#pragma unroll
    for (int tm = 0; tm < 2; ++tm) {
      const __bf16* ap = &lA[(wr + tm * 16 + mn) * 40];
      af[tm] = pack16(*(const v8bf*)(ap + hl * 8), *(const v8bf*)(ap + 16 + hl * 8));
    }
    // B fragments: lane col = mn, elems K = {0..7,16..23}+8*hl (transposed LDS)
    v16bf bfv[4];
#pragma unroll
    for (int tn = 0; tn < 4; ++tn) {
      const __bf16* bp = &lB[(wc + tn * 16 + mn) * 40 + hl * 8];
      bfv[tn] = pack16(*(const v8bf*)bp, *(const v8bf*)(bp + 16));
    }
#pragma unroll
    for (int tm = 0; tm < 2; ++tm)
#pragma unroll
      for (int tn = 0; tn < 4; ++tn)
        acc[tm][tn] = wmma_bf(af[tm], bfv[tn], acc[tm][tn]);
  }
#pragma unroll
  for (int tm = 0; tm < 2; ++tm) {
    const int rrb = bm + wr + tm * 16;
#pragma unroll
    for (int tn = 0; tn < 4; ++tn) {
      const int cc = bn + wc + tn * 16 + mn;
      const float bv = bias[cc];
#pragma unroll
      for (int r = 0; r < 8; ++r) {
        const int rr = rrb + r + 8 * hl;  // C-layout: M = r + 8*(lane/16)
        float v = acc[tm][tn][r] + bv;
        if constexpr (RESID) v += resid[(size_t)rr * Nn + cc];
        if constexpr (ACT == 1) v = fmaxf(v, 0.0f);
        if constexpr (WC) C[(size_t)rr * Nn + cc] = v;
        if constexpr (WBF) Cbf[(size_t)rr * Nn + cc] = (__bf16)v;
      }
    }
  }
}

// ---------------- GAT init: out = bias (broadcast), smax=-inf, den=0 ----------------
__global__ void k_gat_init(float* __restrict__ gat, const float* __restrict__ bias,
                           unsigned* __restrict__ smax, float* __restrict__ den) {
  int i = blockIdx.x * blockDim.x + threadIdx.x;
  gat[i] = bias[i & (HDM - 1)];
  if (i < N_NODES * HG) { smax[i] = FENC_NEG_INF; den[i] = 0.0f; }
}

// ---------------- GAT pass 1: per-edge scores + atomic segment max ----------------
__global__ __launch_bounds__(256) void k_edge_score(
    const float* __restrict__ xl, const float* __restrict__ xr,
    const float* __restrict__ eat, const float* __restrict__ We,
    const float* __restrict__ att, const int* __restrict__ src,
    const int* __restrict__ dst, float* __restrict__ score,
    unsigned* __restrict__ smax) {
  int wid = blockIdx.x * (blockDim.x >> 5) + (threadIdx.x >> 5);
  if (wid >= EE) return;
  int lane = threadIdx.x & 31;
  int s = src[wid], d = dst[wid];
  float ea[EDIM];
#pragma unroll
  for (int j = 0; j < EDIM; ++j) ea[j] = eat[wid * EDIM + j];
  float sc[HG] = {0.f, 0.f, 0.f, 0.f};
  for (int c = lane; c < HDM; c += 32) {
    float v = xl[(size_t)s * HDM + c] + xr[(size_t)d * HDM + c];
#pragma unroll
    for (int j = 0; j < EDIM; ++j) v += ea[j] * We[j * HDM + c];
    v = v > 0.0f ? v : 0.2f * v;              // leaky_relu(0.2)
    sc[c >> 8] += v * att[c];                 // att is [H,D] flat
  }
#pragma unroll
  for (int h = 0; h < HG; ++h)
    for (int m = 16; m >= 1; m >>= 1) sc[h] += __shfl_xor(sc[h], m, 32);
  if (lane < HG) {
    score[wid * HG + lane] = sc[lane];
    atomicMax(&smax[d * HG + lane], fenc(sc[lane]));
  }
}

// ---------------- GAT pass 2: ex = exp(score - max), atomic denominator ----------------
__global__ void k_edge_exp(float* __restrict__ score, const unsigned* __restrict__ smax,
                           float* __restrict__ den, const int* __restrict__ dst,
                           int total) {
  int i = blockIdx.x * blockDim.x + threadIdx.x;
  if (i >= total) return;
  int e = i >> 2, h = i & 3;
  int d = dst[e];
  float ex = __expf(score[i] - fdec(smax[d * HG + h]));
  score[i] = ex;
  atomicAdd(&den[d * HG + h], ex);
}

// ---------------- GAT pass 3: scatter alpha * xl[src] into out[dst] ----------------
__global__ __launch_bounds__(256) void k_edge_scatter(
    const float* __restrict__ xl, const float* __restrict__ score,
    const float* __restrict__ den, const int* __restrict__ src,
    const int* __restrict__ dst, float* __restrict__ gat) {
  int wid = blockIdx.x * (blockDim.x >> 5) + (threadIdx.x >> 5);
  if (wid >= EE) return;
  int lane = threadIdx.x & 31;
  int s = src[wid], d = dst[wid];
  float alpha[HG];
#pragma unroll
  for (int h = 0; h < HG; ++h) alpha[h] = score[wid * HG + h] / den[d * HG + h];
  for (int c = lane; c < HDM; c += 32)
    atomicAdd(&gat[(size_t)d * HDM + c], alpha[c >> 8] * xl[(size_t)s * HDM + c]);
}

// ---------------- flash attention: 2 heads, HD=128, seq N=4096 ----------------
// block = (query-tile 64, head); 4 waves x 16 queries; 32-key streaming tiles
__global__ __launch_bounds__(128) void k_attn(
    const __bf16* __restrict__ qb, const __bf16* __restrict__ kb,
    const __bf16* __restrict__ vb, float* __restrict__ o) {
  const int head = blockIdx.y;
  const int hc = head * 128;
  const int wave = threadIdx.x >> 5, lane = threadIdx.x & 31;
  const int hl = lane >> 4, ln16 = lane & 15;
  const int qrow = blockIdx.x * 64 + wave * 16;

  __shared__ __bf16 lV[128 * 40];      // V tile transposed [channel][key]
  __shared__ __bf16 lP[4][16 * 40];    // per-wave P tile [16 q][32 keys]

  // Q fragments resident in registers (A-layout, 4 chunks of K=32 channels)
  v16bf Qf[4];
#pragma unroll
  for (int kc = 0; kc < 4; ++kc) {
    const __bf16* qp = qb + (size_t)(qrow + ln16) * DM + hc + kc * 32;
    Qf[kc] = pack16(*(const v8bf*)(qp + hl * 8), *(const v8bf*)(qp + 16 + hl * 8));
  }
  v8f accO[8];
  float mi[8], li[8];
#pragma unroll
  for (int ct = 0; ct < 8; ++ct)
#pragma unroll
    for (int r = 0; r < 8; ++r) accO[ct][r] = 0.0f;
#pragma unroll
  for (int r = 0; r < 8; ++r) { mi[r] = -3.0e38f; li[r] = 0.0f; }

  const float scale = 0.08838834764831845f;  // 1/sqrt(128)

  for (int kt = 0; kt < N_NODES; kt += 32) {
    // stage V tile transposed into LDS (all 4 waves cooperate)
    {
      int key = threadIdx.x >> 2, cs = (threadIdx.x & 3) * 32;
      const __bf16* vp = vb + (size_t)(kt + key) * DM + hc + cs;
#pragma unroll
      for (int i = 0; i < 32; ++i) lV[(cs + i) * 40 + key] = vp[i];
    }
    // S = Q K^T (K fragments loaded directly from global: contiguous channels)
    v8f S0, S1;
#pragma unroll
    for (int r = 0; r < 8; ++r) { S0[r] = 0.0f; S1[r] = 0.0f; }
#pragma unroll
    for (int kc = 0; kc < 4; ++kc) {
      const __bf16* kp0 = kb + (size_t)(kt + ln16) * DM + hc + kc * 32 + hl * 8;
      const __bf16* kp1 = kb + (size_t)(kt + 16 + ln16) * DM + hc + kc * 32 + hl * 8;
      v16bf Bf0 = pack16(*(const v8bf*)kp0, *(const v8bf*)(kp0 + 16));
      v16bf Bf1 = pack16(*(const v8bf*)kp1, *(const v8bf*)(kp1 + 16));
      S0 = wmma_bf(Qf[kc], Bf0, S0);
      S1 = wmma_bf(Qf[kc], Bf1, S1);
    }
    // online softmax update (row m = r + 8*hl lives in the 16-lane half)
#pragma unroll
    for (int r = 0; r < 8; ++r) {
      float s0 = S0[r] * scale, s1 = S1[r] * scale;
      float mx = fmaxf(s0, s1);
      for (int m = 8; m >= 1; m >>= 1) mx = fmaxf(mx, __shfl_xor(mx, m, 32));
      float mnew = fmaxf(mi[r], mx);
      float corr = __expf(mi[r] - mnew);
      float p0 = __expf(s0 - mnew);
      float p1 = __expf(s1 - mnew);
      float rs = p0 + p1;
      for (int m = 8; m >= 1; m >>= 1) rs += __shfl_xor(rs, m, 32);
      li[r] = li[r] * corr + rs;
      mi[r] = mnew;
#pragma unroll
      for (int ct = 0; ct < 8; ++ct) accO[ct][r] *= corr;
      __bf16* pp = &lP[wave][(r + 8 * hl) * 40];
      pp[ln16] = (__bf16)p0;
      pp[16 + ln16] = (__bf16)p1;
    }
    __syncthreads();  // covers V staging + P writes
    // P (A-frag from LDS) x V (B-frags from transposed LDS)
    const __bf16* ap = &lP[wave][ln16 * 40];
    v16bf Pf = pack16(*(const v8bf*)(ap + hl * 8), *(const v8bf*)(ap + 16 + hl * 8));
#pragma unroll
    for (int ct = 0; ct < 8; ++ct) {
      const __bf16* vp2 = &lV[(ct * 16 + ln16) * 40 + hl * 8];
      v16bf Vf = pack16(*(const v8bf*)vp2, *(const v8bf*)(vp2 + 16));
      accO[ct] = wmma_bf(Pf, Vf, accO[ct]);
    }
    __syncthreads();  // protect lV/lP before next iteration
  }
#pragma unroll
  for (int r = 0; r < 8; ++r) {
    float inv = 1.0f / li[r];
    int rr = qrow + r + 8 * hl;
#pragma unroll
    for (int ct = 0; ct < 8; ++ct)
      o[(size_t)rr * DM + hc + ct * 16 + ln16] = accO[ct][r] * inv;
  }
}

// ---------------- global attention pooling (segments are contiguous S=128 blocks) ----
__global__ __launch_bounds__(128) void k_pool(
    const float* __restrict__ hn, const float* __restrict__ gw,
    const float* __restrict__ gb, float* __restrict__ pooled) {
  int g = blockIdx.x, t = threadIdx.x;
  const float* hr = hn + (size_t)(g * SN + t) * DM;
  float gate = gb[0];
  for (int j = 0; j < DM; ++j) gate += hr[j] * gw[j];
  __shared__ float red[SN];
  __shared__ float aw[SN];
  red[t] = gate; __syncthreads();
  for (int s = 64; s > 0; s >>= 1) { if (t < s) red[t] = fmaxf(red[t], red[t + s]); __syncthreads(); }
  float gm = red[0]; __syncthreads();
  float ex = __expf(gate - gm);
  red[t] = ex; __syncthreads();
  for (int s = 64; s > 0; s >>= 1) { if (t < s) red[t] += red[t + s]; __syncthreads(); }
  aw[t] = ex / red[0]; __syncthreads();
  for (int d = t; d < DM; d += SN) {
    float acc = 0.0f;
    for (int n2 = 0; n2 < SN; ++n2) acc += aw[n2] * hn[(size_t)(g * SN + n2) * DM + d];
    pooled[g * DM + d] = acc;
  }
}

// ---------------- classifier head (per graph) ----------------
__global__ __launch_bounds__(128) void k_head(
    const float* __restrict__ pooled, const float* __restrict__ c1w,
    const float* __restrict__ c1b, const float* __restrict__ c1g,
    const float* __restrict__ c1bb, const float* __restrict__ rbw,
    const float* __restrict__ rbb2, const float* __restrict__ rbg,
    const float* __restrict__ rbbb, const float* __restrict__ c2w,
    const float* __restrict__ c2b, float* __restrict__ out) {
  int g = blockIdx.x, t = threadIdx.x;
  __shared__ float red[128];
  __shared__ float zs[128];
  const float* pr = pooled + g * DM;
  float z = c1b[t];
  for (int j = 0; j < DM; ++j) z += pr[j] * c1w[j * 128 + t];
  red[t] = z; __syncthreads();
  for (int s = 64; s > 0; s >>= 1) { if (t < s) red[t] += red[t + s]; __syncthreads(); }
  float mean = red[0] * (1.0f / 128.0f); __syncthreads();
  float d = z - mean; red[t] = d * d; __syncthreads();
  for (int s = 64; s > 0; s >>= 1) { if (t < s) red[t] += red[t + s]; __syncthreads(); }
  float var = red[0] * (1.0f / 128.0f); __syncthreads();
  z = gelu_f(d * rsqrtf(var + 1e-5f) * c1g[t] + c1bb[t]);
  zs[t] = z; __syncthreads();
  float r = rbb2[t];
  for (int j = 0; j < 128; ++j) r += zs[j] * rbw[j * 128 + t];
  red[t] = r; __syncthreads();
  for (int s = 64; s > 0; s >>= 1) { if (t < s) red[t] += red[t + s]; __syncthreads(); }
  float mean2 = red[0] * (1.0f / 128.0f); __syncthreads();
  float d2 = r - mean2; red[t] = d2 * d2; __syncthreads();
  for (int s = 64; s > 0; s >>= 1) { if (t < s) red[t] += red[t + s]; __syncthreads(); }
  float var2 = red[0] * (1.0f / 128.0f); __syncthreads();
  float rn = d2 * rsqrtf(var2 + 1e-5f) * rbg[t] + rbbb[t];
  float z2 = z + gelu_f(rn);
  red[t] = z2 * c2w[t]; __syncthreads();
  for (int s = 64; s > 0; s >>= 1) { if (t < s) red[t] += red[t + s]; __syncthreads(); }
  if (t == 0) out[g] = red[0] + c2b[0];
}

// ---------------- host orchestration ----------------
extern "C" void kernel_launch(void* const* d_in, const int* in_sizes, int n_in,
                              void* d_out, int out_size, void* d_ws, size_t ws_size,
                              hipStream_t stream) {
  auto F = [&](int i) { return (const float*)d_in[i]; };
  const float* x      = F(0);
  const int*   ei     = (const int*)d_in[1];
  const float* eattr  = F(2);
  // d_in[3] = batch (structure known: node n -> graph n/128)
  const float* emb_w  = F(4);
  const float* emb_b  = F(5);
  const float* emb_g  = F(6);
  const float* emb_bb = F(7);
  const float* pos    = F(8);
  struct Lp { const float *Wl,*bl,*Wr,*br,*We,*att,*gbias,*Wt,*bt,*Wq,*bq,*Wk,*bk,*Wv,*bv,*Wo,*bo; };
  Lp L[NLAY];
  int idx = 9;
  for (int l = 0; l < NLAY; ++l) {
    L[l].Wl = F(idx++); L[l].bl = F(idx++); L[l].Wr = F(idx++); L[l].br = F(idx++);
    L[l].We = F(idx++); L[l].att = F(idx++); L[l].gbias = F(idx++);
    L[l].Wt = F(idx++); L[l].bt = F(idx++);
    L[l].Wq = F(idx++); L[l].bq = F(idx++); L[l].Wk = F(idx++); L[l].bk = F(idx++);
    L[l].Wv = F(idx++); L[l].bv = F(idx++); L[l].Wo = F(idx++); L[l].bo = F(idx++);
  }
  const float* pre_g = F(idx++); const float* pre_b = F(idx++);
  const float* gatew = F(idx++); const float* gateb = F(idx++);
  const float* c1w = F(idx++);  const float* c1b = F(idx++);
  const float* c1g = F(idx++);  const float* c1bb = F(idx++);
  const float* rbw = F(idx++);  const float* rbb = F(idx++);
  const float* rbg = F(idx++);  const float* rbbb = F(idx++);
  const float* c2w = F(idx++);  const float* c2b = F(idx++);

  const int* src = ei;
  const int* dst = ei + EE;

  // workspace carve (bump allocator, 256B aligned)
  char* wp = (char*)d_ws;
  auto alloc = [&](size_t bytes) {
    void* r = (void*)wp; wp += (bytes + 255) & ~(size_t)255; return r;
  };
  float*   h      = (float*)alloc(sizeof(float) * N_NODES * DM);
  __bf16*  hbf    = (__bf16*)alloc(sizeof(__bf16) * N_NODES * DM);
  float*   hn     = (float*)alloc(sizeof(float) * N_NODES * DM);
  float*   xl     = (float*)alloc(sizeof(float) * (size_t)N_NODES * HDM);
  float*   xr     = (float*)alloc(sizeof(float) * (size_t)N_NODES * HDM);
  float*   gat    = (float*)alloc(sizeof(float) * (size_t)N_NODES * HDM);
  __bf16*  gatbf  = (__bf16*)alloc(sizeof(__bf16) * (size_t)N_NODES * HDM);
  float*   h2     = (float*)alloc(sizeof(float) * N_NODES * DM);
  __bf16*  h2bf   = (__bf16*)alloc(sizeof(__bf16) * N_NODES * DM);
  __bf16*  qbf    = (__bf16*)alloc(sizeof(__bf16) * N_NODES * DM);
  __bf16*  kbf    = (__bf16*)alloc(sizeof(__bf16) * N_NODES * DM);
  __bf16*  vbf    = (__bf16*)alloc(sizeof(__bf16) * N_NODES * DM);
  float*   oat    = (float*)alloc(sizeof(float) * N_NODES * DM);
  __bf16*  obf    = (__bf16*)alloc(sizeof(__bf16) * N_NODES * DM);
  float*   score  = (float*)alloc(sizeof(float) * EE * HG);
  unsigned* smax  = (unsigned*)alloc(sizeof(unsigned) * N_NODES * HG);
  float*   den    = (float*)alloc(sizeof(float) * N_NODES * HG);
  __bf16*  wbuf   = (__bf16*)alloc(sizeof(__bf16) * 256 * 1024);
  float*   pooled = (float*)alloc(sizeof(float) * GN * DM);

  // ---- embedding + pos enc ----
  k_embed<<<N_NODES, 256, 0, stream>>>(x, emb_w, emb_b, emb_g, emb_bb, pos, h, hbf);

  for (int l = 0; l < NLAY; ++l) {
    // xl = h @ Wl + bl   [4096x256 x 256x1024]
    k_cvt<<<512, 256, 0, stream>>>(L[l].Wl, wbuf, DM * HDM);
    k_gemm<0, false, true, false><<<dim3(HDM / 128, N_NODES / 128), 256, 0, stream>>>(
        hbf, wbuf, L[l].bl, nullptr, xl, nullptr, N_NODES, HDM, DM);
    // xr = h @ Wr + br
    k_cvt<<<512, 256, 0, stream>>>(L[l].Wr, wbuf, DM * HDM);
    k_gemm<0, false, true, false><<<dim3(HDM / 128, N_NODES / 128), 256, 0, stream>>>(
        hbf, wbuf, L[l].br, nullptr, xr, nullptr, N_NODES, HDM, DM);
    // GATv2 edge softmax + scatter
    k_gat_init<<<(N_NODES * HDM) / 256, 256, 0, stream>>>(gat, L[l].gbias, smax, den);
    k_edge_score<<<EE / 8, 256, 0, stream>>>(xl, xr, eattr, L[l].We, L[l].att,
                                             src, dst, score, smax);
    k_edge_exp<<<(EE * HG) / 256, 256, 0, stream>>>(score, smax, den, dst, EE * HG);
    k_edge_scatter<<<EE / 8, 256, 0, stream>>>(xl, score, den, src, dst, gat);
    // h2 = relu(gat @ Wt + bt)   [4096x1024 x 1024x256]
    k_cvt<<<1024, 256, 0, stream>>>(gat, gatbf, N_NODES * HDM);
    k_cvt<<<512, 256, 0, stream>>>(L[l].Wt, wbuf, HDM * DM);
    k_gemm<1, false, true, true><<<dim3(DM / 128, N_NODES / 128), 256, 0, stream>>>(
        gatbf, wbuf, L[l].bt, nullptr, h2, h2bf, N_NODES, DM, HDM);
    // q,k,v projections (bf16 outputs only)
    k_cvt<<<256, 256, 0, stream>>>(L[l].Wq, wbuf, DM * DM);
    k_gemm<0, false, false, true><<<dim3(DM / 128, N_NODES / 128), 256, 0, stream>>>(
        h2bf, wbuf, L[l].bq, nullptr, nullptr, qbf, N_NODES, DM, DM);
    k_cvt<<<256, 256, 0, stream>>>(L[l].Wk, wbuf, DM * DM);
    k_gemm<0, false, false, true><<<dim3(DM / 128, N_NODES / 128), 256, 0, stream>>>(
        h2bf, wbuf, L[l].bk, nullptr, nullptr, kbf, N_NODES, DM, DM);
    k_cvt<<<256, 256, 0, stream>>>(L[l].Wv, wbuf, DM * DM);
    k_gemm<0, false, false, true><<<dim3(DM / 128, N_NODES / 128), 256, 0, stream>>>(
        h2bf, wbuf, L[l].bv, nullptr, nullptr, vbf, N_NODES, DM, DM);
    // flash attention over full node sequence, 2 heads
    k_attn<<<dim3(N_NODES / 64, 2), 128, 0, stream>>>(qbf, kbf, vbf, oat);
    // h = o @ Wo + bo + residual(h)
    k_cvt<<<1024, 256, 0, stream>>>(oat, obf, N_NODES * DM);
    k_cvt<<<256, 256, 0, stream>>>(L[l].Wo, wbuf, DM * DM);
    k_gemm<0, true, true, true><<<dim3(DM / 128, N_NODES / 128), 256, 0, stream>>>(
        obf, wbuf, L[l].bo, h, h, hbf, N_NODES, DM, DM);
  }

  // final LN, pooling, classifier
  k_ln<<<N_NODES, 256, 0, stream>>>(h, pre_g, pre_b, hn);
  k_pool<<<GN, 128, 0, stream>>>(hn, gatew, gateb, pooled);
  k_head<<<GN, 128, 0, stream>>>(pooled, c1w, c1b, c1g, c1bb, rbw, rbb, rbg, rbbb,
                                 c2w, c2b, (float*)d_out);
  (void)in_sizes; (void)n_in; (void)out_size; (void)ws_size;
}
